// MMDWithImWeights_73907797230167
// MI455X (gfx1250) — matrix operand order repaired and checked by reading, not compiled, not run on hardware
//
#include <hip/hip_runtime.h>

typedef __attribute__((ext_vector_type(16))) __bf16 v16bf;
typedef __attribute__((ext_vector_type(8)))  float  v8f;

union Frag { v16bf v; uint4 q[2]; };

static __device__ __forceinline__ v8f wmma_bf16(v16bf a, v16bf b, v8f c) {
  return __builtin_amdgcn_wmma_f32_16x16x32_bf16(false, a, false, b, (short)0, c, false, false);
}

// ---------------------------------------------------------------------------
// Kernel 1: split x into bf16 hi/lo, compute per-row squared norms (fp32)
// ---------------------------------------------------------------------------
__global__ void prep_kernel(const float* __restrict__ x,
                            __bf16* __restrict__ Xhi, __bf16* __restrict__ Xlo,
                            float* __restrict__ sq, int D) {
  const int row = blockIdx.x;
  const float* xr = x + (size_t)row * D;
  float acc = 0.f;
  for (int c = threadIdx.x; c < D; c += blockDim.x) {
    float v = xr[c];
    __bf16 h = (__bf16)v;
    __bf16 l = (__bf16)(v - (float)h);
    Xhi[(size_t)row * D + c] = h;
    Xlo[(size_t)row * D + c] = l;
    acc += v * v;
  }
  __shared__ float red[128];
  red[threadIdx.x] = acc;
  __syncthreads();
  for (int s = 64; s > 0; s >>= 1) {
    if (threadIdx.x < s) red[threadIdx.x] += red[threadIdx.x + s];
    __syncthreads();
  }
  if (threadIdx.x == 0) sq[row] = red[0];
}

// ---------------------------------------------------------------------------
// Kernel 2: column sums s[d] = sum_i x[i,d] (for analytic bandwidth)
// ---------------------------------------------------------------------------
__global__ void colsum_kernel(const float* __restrict__ x, float* __restrict__ s,
                              int n, int D) {
  int d = blockIdx.x * blockDim.x + threadIdx.x;
  if (d >= D) return;
  double acc = 0.0;
  for (int i = 0; i < n; ++i) acc += (double)x[(size_t)i * D + d];
  s[d] = (float)acc;
}

// ---------------------------------------------------------------------------
// Kernel 3: bandwidth scalar + per-source-sample weights
//   sum(l2) = 2*N*sum(sq) - 2*||colsum||^2 ; bw = sum/(N^2-N)/KERNEL_MUL^2
// ---------------------------------------------------------------------------
__global__ void scalars_kernel(const float* __restrict__ sq, const float* __restrict__ s,
                               float* __restrict__ scal, float* __restrict__ wsrc,
                               const float* __restrict__ imw, const int* __restrict__ y_true,
                               const int* __restrict__ alpha_i, int n, int D, int ns) {
  __shared__ double red[256];
  double a = 0.0;
  for (int i = threadIdx.x; i < n; i += 256) a += (double)sq[i];
  red[threadIdx.x] = a; __syncthreads();
  for (int st = 128; st > 0; st >>= 1) {
    if (threadIdx.x < st) red[threadIdx.x] += red[threadIdx.x + st];
    __syncthreads();
  }
  double sumsq = red[0];
  __syncthreads();
  double b = 0.0;
  for (int d = threadIdx.x; d < D; d += 256) { double v = (double)s[d]; b += v * v; }
  red[threadIdx.x] = b; __syncthreads();
  for (int st = 128; st > 0; st >>= 1) {
    if (threadIdx.x < st) red[threadIdx.x] += red[threadIdx.x + st];
    __syncthreads();
  }
  double ssq = red[0];
  if (threadIdx.x == 0) {
    double suml2 = 2.0 * (double)n * sumsq - 2.0 * ssq;
    double denom = (double)n * (double)n - (double)n;
    double bw = (suml2 / denom) / 4.0;       // / KERNEL_MUL^(KERNEL_NUM/2) = 2^2
    scal[0] = (float)(-1.0 / bw);            // exponent coefficient for t=0
  }
  float alpha = (float)alpha_i[0];
  for (int i = threadIdx.x; i < ns; i += 256)
    wsrc[i] = alpha * imw[y_true[i]] + (1.f - alpha);
}

// ---------------------------------------------------------------------------
// Kernel 4: fused Gram(WMMA bf16 hi/lo) -> l2 -> 5x exp -> weighted tile sum
// grid = (n/64, n/128), 256 threads = 8 waves; wave owns 32x32 (4 WMMA tiles)
// ---------------------------------------------------------------------------
static __device__ __forceinline__ float tile_contrib(
    v8f acc, int rbase, int cbase, int lane,
    const float* __restrict__ sq, const float* __restrict__ wsrc,
    float c0, int ns, float invSS, float invTT, float invST) {
  const int m  = lane & 15;
  const int hb = lane >> 4;
  const int col = cbase + m;
  const float sqc = sq[col];
  const float wc  = (col < ns) ? wsrc[col] : 0.f;
  float sum = 0.f;
#pragma unroll
  for (int r = 0; r < 8; ++r) {
    const int row = rbase + hb * 8 + r;
    float g  = acc[r];
    float l2 = fmaxf(sq[row] + sqc - 2.f * g, 0.f);
    float c = c0, k = 0.f;
#pragma unroll
    for (int t = 0; t < 5; ++t) { k += __expf(l2 * c); c *= 0.5f; }
    float coef;
    if (row == col) {
      coef = 0.f;                                           // triu(k=1): skip diagonal
    } else if (row < ns) {
      coef = (col < ns) ? (wsrc[row] * wc * invSS)          // SS, halved into off-diag
                        : (-wsrc[row] * invST);             // ST (-2*loss2, mirrored)
    } else {
      coef = (col < ns) ? (-wc * invST)                     // TS mirror
                        : invTT;                            // TT
    }
    sum += k * coef;
  }
  return sum;
}

__global__ void mmd_kernel(const __bf16* __restrict__ Xhi, const __bf16* __restrict__ Xlo,
                           const float* __restrict__ sq, const float* __restrict__ wsrc,
                           const float* __restrict__ scal, double* __restrict__ partials,
                           int n, int D, int ns,
                           float invSS, float invTT, float invST) {
  const int lane = threadIdx.x & 31;
  const int wave = threadIdx.x >> 5;
  const int tr = blockIdx.y * 128 + (wave >> 1) * 32;   // wave's 32 rows
  const int tc = blockIdx.x * 64  + (wave & 1) * 32;    // wave's 32 cols
  const int m  = lane & 15;
  const int hb = lane >> 4;

  // A (16x32 bf16): lane m holds K = hb*8 + [0..7] and +16..+23  -> two b128 loads
  // B (32x16 bf16): lane m (col) holds K = hb*8 + [0..7] and +16..+23 -> two b128 loads
  const size_t aB0 = (size_t)(tr + m) * D + (hb << 3);
  const size_t aB1 = aB0 + (size_t)16 * D;
  const size_t bB0 = (size_t)(tc + m) * D + (hb << 3);
  const size_t bB1 = bB0 + (size_t)16 * D;

  v8f acc00 = {}, acc01 = {}, acc10 = {}, acc11 = {};

  for (int kk = 0; kk < D; kk += 32) {
    Frag Ah0, Al0, Ah1, Al1, Bh0, Bl0, Bh1, Bl1;
    Ah0.q[0] = *(const uint4*)(Xhi + aB0 + kk);
    Ah0.q[1] = *(const uint4*)(Xhi + aB0 + kk + 16);
    Al0.q[0] = *(const uint4*)(Xlo + aB0 + kk);
    Al0.q[1] = *(const uint4*)(Xlo + aB0 + kk + 16);
    Ah1.q[0] = *(const uint4*)(Xhi + aB1 + kk);
    Ah1.q[1] = *(const uint4*)(Xhi + aB1 + kk + 16);
    Al1.q[0] = *(const uint4*)(Xlo + aB1 + kk);
    Al1.q[1] = *(const uint4*)(Xlo + aB1 + kk + 16);
    Bh0.q[0] = *(const uint4*)(Xhi + bB0 + kk);
    Bh0.q[1] = *(const uint4*)(Xhi + bB0 + kk + 16);
    Bl0.q[0] = *(const uint4*)(Xlo + bB0 + kk);
    Bl0.q[1] = *(const uint4*)(Xlo + bB0 + kk + 16);
    Bh1.q[0] = *(const uint4*)(Xhi + bB1 + kk);
    Bh1.q[1] = *(const uint4*)(Xhi + bB1 + kk + 16);
    Bl1.q[0] = *(const uint4*)(Xlo + bB1 + kk);
    Bl1.q[1] = *(const uint4*)(Xlo + bB1 + kk + 16);

    // G = hi*hi + hi*lo + lo*hi  (lo*lo ~ 2^-18, dropped)
    acc00 = wmma_bf16(Ah0.v, Bh0.v, acc00);
    acc00 = wmma_bf16(Ah0.v, Bl0.v, acc00);
    acc00 = wmma_bf16(Al0.v, Bh0.v, acc00);

    acc01 = wmma_bf16(Ah0.v, Bh1.v, acc01);
    acc01 = wmma_bf16(Ah0.v, Bl1.v, acc01);
    acc01 = wmma_bf16(Al0.v, Bh1.v, acc01);

    acc10 = wmma_bf16(Ah1.v, Bh0.v, acc10);
    acc10 = wmma_bf16(Ah1.v, Bl0.v, acc10);
    acc10 = wmma_bf16(Al1.v, Bh0.v, acc10);

    acc11 = wmma_bf16(Ah1.v, Bh1.v, acc11);
    acc11 = wmma_bf16(Ah1.v, Bl1.v, acc11);
    acc11 = wmma_bf16(Al1.v, Bh1.v, acc11);
  }

  const float c0 = scal[0];
  float lsum = 0.f;
  lsum += tile_contrib(acc00, tr,      tc,      lane, sq, wsrc, c0, ns, invSS, invTT, invST);
  lsum += tile_contrib(acc01, tr,      tc + 16, lane, sq, wsrc, c0, ns, invSS, invTT, invST);
  lsum += tile_contrib(acc10, tr + 16, tc,      lane, sq, wsrc, c0, ns, invSS, invTT, invST);
  lsum += tile_contrib(acc11, tr + 16, tc + 16, lane, sq, wsrc, c0, ns, invSS, invTT, invST);

  __shared__ float red[256];
  red[threadIdx.x] = lsum;
  __syncthreads();
  for (int s = 128; s > 0; s >>= 1) {
    if (threadIdx.x < s) red[threadIdx.x] += red[threadIdx.x + s];
    __syncthreads();
  }
  if (threadIdx.x == 0)
    partials[(size_t)blockIdx.y * gridDim.x + blockIdx.x] = (double)red[0];
}

// ---------------------------------------------------------------------------
// Kernel 5: deterministic fixed-order final reduction of tile partials
// ---------------------------------------------------------------------------
__global__ void finish_kernel(const double* __restrict__ partials, int nTiles,
                              float* __restrict__ out) {
  __shared__ double red[256];
  double a = 0.0;
  for (int i = threadIdx.x; i < nTiles; i += 256) a += partials[i];
  red[threadIdx.x] = a; __syncthreads();
  for (int s = 128; s > 0; s >>= 1) {
    if (threadIdx.x < s) red[threadIdx.x] += red[threadIdx.x + s];
    __syncthreads();
  }
  if (threadIdx.x == 0) out[0] = (float)red[0];
}

// ---------------------------------------------------------------------------
extern "C" void kernel_launch(void* const* d_in, const int* in_sizes, int n_in,
                              void* d_out, int out_size, void* d_ws, size_t ws_size,
                              hipStream_t stream) {
  const float* x      = (const float*)d_in[0];
  const float* imw    = (const float*)d_in[1];
  // d_in[2] = domain_labels: setup guarantees [0]*ns ++ [1]*nt
  const int*   y_true = (const int*)d_in[3];
  const int*   alpha  = (const int*)d_in[4];

  const int n  = in_sizes[2];           // 8192
  const int ns = in_sizes[3];           // 4096
  const int nt = n - ns;                // 4096
  const int D  = in_sizes[0] / n;       // 512

  char* ws = (char*)d_ws;
  size_t off = 0;
  auto alloc = [&](size_t bytes) -> void* {
    void* p = ws + off;
    off += (bytes + 255) & ~(size_t)255;
    return p;
  };
  __bf16* Xhi      = (__bf16*)alloc((size_t)n * D * sizeof(__bf16));
  __bf16* Xlo      = (__bf16*)alloc((size_t)n * D * sizeof(__bf16));
  float*  sq       = (float*) alloc((size_t)n * sizeof(float));
  float*  s        = (float*) alloc((size_t)D * sizeof(float));
  float*  wsrc     = (float*) alloc((size_t)ns * sizeof(float));
  float*  scal     = (float*) alloc(64);
  const int tilesX = n / 64, tilesY = n / 128;
  const int nTiles = tilesX * tilesY;
  double* partials = (double*)alloc((size_t)nTiles * sizeof(double));

  const float invSS = 1.0f / ((float)ns * (float)(ns - 1));
  const float invTT = 1.0f / ((float)nt * (float)(nt - 1));
  const float invST = 1.0f / ((float)ns * (float)nt);

  prep_kernel<<<n, 128, 0, stream>>>(x, Xhi, Xlo, sq, D);
  colsum_kernel<<<(D + 255) / 256, 256, 0, stream>>>(x, s, n, D);
  scalars_kernel<<<1, 256, 0, stream>>>(sq, s, scal, wsrc, imw, y_true, alpha, n, D, ns);
  mmd_kernel<<<dim3(tilesX, tilesY), 256, 0, stream>>>(Xhi, Xlo, sq, wsrc, scal, partials,
                                                       n, D, ns, invSS, invTT, invST);
  finish_kernel<<<1, 256, 0, stream>>>(partials, nTiles, (float*)d_out);
}
